// HeterogeneousGAT_28527172780181
// MI455X (gfx1250) — matrix operand order repaired and checked by reading, not compiled, not run on hardware
//
#include <hip/hip_runtime.h>
#include <hip/hip_bf16.h>

// ---------------------------------------------------------------------------
// CDNA5 (gfx1250) fused GNN implementation.
// Heavy math runs through v_wmma_f32_16x16x32_bf16 (bf16 in, f32 accum).
// Persistent grid-stride blocks: weights staged in LDS once per block.
// Kernel templated on <MODE, DIN> so A-fragment gathers are wide b128/b64
// loads with no per-element EXEC-masked guards.
// ---------------------------------------------------------------------------

typedef __attribute__((ext_vector_type(16))) __bf16       v16bf;
typedef __attribute__((ext_vector_type(8)))  float        v8f;
typedef __attribute__((ext_vector_type(4)))  unsigned int u32x4;
typedef __attribute__((ext_vector_type(4)))  float        f32x4;
typedef __attribute__((ext_vector_type(2)))  float        f32x2;

union BFV {
  v16bf v;
  __bf16 h[16];
  unsigned short u[16];
  u32x4 q[2];
};

__device__ __forceinline__ unsigned short f2bf(float f) {
  // native f32 -> bf16 (RNE); lowers to v_cvt_pk_bf16_f32 on gfx1250
  return __builtin_bit_cast(unsigned short, static_cast<__bf16>(f));
}
__device__ __forceinline__ float bf2f(unsigned short s) {
  return __uint_as_float(((unsigned int)s) << 16);
}
__device__ __forceinline__ float elu1(float x) {
  return x > 0.f ? x : (__expf(x) - 1.f);
}

// ---------------------------------------------------------------------------
// Generic fused MLP (DIN -> 128 -> 128 -> 8), 16 rows per wave-tile.
// MODE 0: dense rows from in_x [nrows, DIN], dense output out[row][8]
// MODE 1: per-edge rows; input = concat[g0[dst], g1[src], g2[dst]] (24 dims),
//         output = mlp(...) + addv[dst], atomicAdd into out[dst][8]
// Weight layouts (pre-transposed by prep kernel):
//   W0t bf16 [128][32]  (n-major, k zero-padded to 32)  -> register-resident B
//   W1t bf16 [128][128] (n-major)                        -> LDS-resident B
//   W2t f32  [8][128]   (o-major)                        -> LDS
// ---------------------------------------------------------------------------
#define MLP_WAVES      4
#define MLP_BLOCK      (MLP_WAVES * 32)
#define ROWS_PER_BLOCK (MLP_WAVES * 16)
#define STG            136   // bf16 stage row stride (16B-aligned rows, bank-skewed)

template <int MODE, int DIN>
__global__ __launch_bounds__(MLP_BLOCK)
void mlp16_kernel(const float* __restrict__ in_x,
                  const float* __restrict__ g0, const float* __restrict__ g1,
                  const float* __restrict__ g2,
                  const int* __restrict__ e_src, const int* __restrict__ e_dst,
                  const float* __restrict__ addv,
                  int nrows, int ntiles,
                  const unsigned short* __restrict__ W0t, const float* __restrict__ b0,
                  const unsigned short* __restrict__ W1t, const float* __restrict__ b1,
                  const float* __restrict__ W2t, const float* __restrict__ b2,
                  float* __restrict__ out)
{
  __shared__ alignas(16) unsigned short sW1t[128 * 128];
  __shared__ alignas(16) float          sW2t[8 * 128];
  __shared__ float sb0[128];
  __shared__ float sb1[128];
  __shared__ float sb2[8];
  __shared__ alignas(16) unsigned short sStage[MLP_WAVES][16 * STG];

  const int tid = threadIdx.x;
  for (int i = tid; i < 128 * 128; i += MLP_BLOCK) sW1t[i] = W1t[i];
  for (int i = tid; i < 8 * 128; i += MLP_BLOCK)   sW2t[i] = W2t[i];
  for (int i = tid; i < 128; i += MLP_BLOCK) { sb0[i] = b0[i]; sb1[i] = b1[i]; }
  if (tid < 8) sb2[tid] = b2[tid];
  __syncthreads();

  const int wave = tid >> 5;
  const int lane = tid & 31;
  const int half = lane >> 4;     // which 16-lane group
  const int mrow = lane & 15;     // M index within the 16-row tile

  // Loop-invariant layer0 B fragments (CDNA5 16-bit B layout: lane holds
  // column n = nt*16+mrow, K = 8*half..+7 then 16+8*half..+7) -> registers.
  BFV b0f[8];
#pragma unroll
  for (int nt = 0; nt < 8; ++nt) {
    const u32x4* p = (const u32x4*)&W0t[(nt * 16 + mrow) * 32 + 8 * half];
    b0f[nt].q[0] = p[0];
    b0f[nt].q[1] = p[2];
  }

  unsigned short* stg = sStage[wave];

  for (int tile = blockIdx.x; tile < ntiles; tile += gridDim.x) {
    const int  row   = tile * ROWS_PER_BLOCK + wave * 16 + mrow;
    const bool valid = row < nrows;
    const int  rc    = valid ? row : (nrows - 1);

    // ---- build A0 in the CDNA5 16-bit A-matrix lane layout (16x32) ----
    // vector element e -> K = (e<8) ? 8*half+e : 16+8*half+(e-8)
    BFV a0;
    int dst = rc;
    if (MODE == 1) {
      const int src = e_src[rc];
      dst = e_dst[rc];
      // half 0: e<8 -> K=0..7  = g0[dst];  e>=8 -> K=16..23 = g2[dst]
      // half 1: e<8 -> K=8..15 = g1[src];  e>=8 -> K=24..31 = 0
      f32x4 lo0, lo1, hi0, hi1;
      if (half == 0) {
        const f32x4* p0 = (const f32x4*)&g0[dst * 8];
        const f32x4* p2 = (const f32x4*)&g2[dst * 8];
        lo0 = p0[0]; lo1 = p0[1];
        hi0 = p2[0]; hi1 = p2[1];
      } else {
        const f32x4* p1 = (const f32x4*)&g1[src * 8];
        lo0 = p1[0]; lo1 = p1[1];
        hi0 = (f32x4){0.f, 0.f, 0.f, 0.f};
        hi1 = (f32x4){0.f, 0.f, 0.f, 0.f};
      }
#pragma unroll
      for (int j = 0; j < 4; ++j) {
        a0.h[j]      = static_cast<__bf16>(lo0[j]);
        a0.h[4 + j]  = static_cast<__bf16>(lo1[j]);
        a0.h[8 + j]  = static_cast<__bf16>(hi0[j]);
        a0.h[12 + j] = static_cast<__bf16>(hi1[j]);
      }
    } else {
      // node mode: only half 0, e<8 carries data (K = 0..DIN-1); rest zero.
      float xv[8];
      if (DIN == 8) {
        const f32x4* p = (const f32x4*)&in_x[(long)rc * 8];
        const f32x4 v0 = p[0], v1 = p[1];
#pragma unroll
        for (int j = 0; j < 4; ++j) { xv[j] = v0[j]; xv[4 + j] = v1[j]; }
      } else {  // DIN == 6
        const f32x2* p = (const f32x2*)&in_x[(long)rc * 6];
        const f32x2 v0 = p[0], v1 = p[1], v2 = p[2];
        xv[0] = v0[0]; xv[1] = v0[1]; xv[2] = v1[0]; xv[3] = v1[1];
        xv[4] = v2[0]; xv[5] = v2[1]; xv[6] = 0.f; xv[7] = 0.f;
      }
#pragma unroll
      for (int e = 0; e < 8; ++e) {
        a0.h[e]     = static_cast<__bf16>(half == 0 ? xv[e] : 0.f);
        a0.h[8 + e] = static_cast<__bf16>(0.f);
      }
    }

    // ---- layer0: [16x32] @ [32x128] = 8 WMMA; ELU; stage h0 (bf16) ----
#pragma unroll
    for (int nt = 0; nt < 8; ++nt) {
      const int n = nt * 16 + mrow;
      v8f c = {0.f, 0.f, 0.f, 0.f, 0.f, 0.f, 0.f, 0.f};
      c = __builtin_amdgcn_wmma_f32_16x16x32_bf16(false, a0.v, false, b0f[nt].v,
                                                  (short)0, c, false, false);
      const float bias = sb0[n];
#pragma unroll
      for (int r = 0; r < 8; ++r) {     // D: lane holds (m = 8*half+r, n)
        const int m = 8 * half + r;
        stg[m * STG + n] = f2bf(elu1(c[r] + bias));
      }
    }
    __syncthreads();

    // ---- reload h0 as four A-format K-chunks ----
    BFV a1[4];
#pragma unroll
    for (int kc = 0; kc < 4; ++kc) {
      const u32x4* p0 = (const u32x4*)&stg[mrow * STG + kc * 32 + 8 * half];
      const u32x4* p1 = (const u32x4*)&stg[mrow * STG + kc * 32 + 16 + 8 * half];
      a1[kc].q[0] = p0[0];
      a1[kc].q[1] = p1[0];
    }

    // ---- layer1: [16x128] @ [128x128] = 8x4 WMMA; ELU; stage h1 (bf16) ----
#pragma unroll
    for (int nt = 0; nt < 8; ++nt) {
      const int n = nt * 16 + mrow;
      v8f c = {0.f, 0.f, 0.f, 0.f, 0.f, 0.f, 0.f, 0.f};
#pragma unroll
      for (int kc = 0; kc < 4; ++kc) {
        BFV b;
        const u32x4* bp = (const u32x4*)&sW1t[n * 128 + kc * 32 + 8 * half];
        b.q[0] = bp[0];
        b.q[1] = bp[2];
        c = __builtin_amdgcn_wmma_f32_16x16x32_bf16(false, a1[kc].v, false, b.v,
                                                    (short)0, c, false, false);
      }
      const float bias = sb1[n];
#pragma unroll
      for (int r = 0; r < 8; ++r) {
        const int m = 8 * half + r;
        stg[m * STG + n] = f2bf(elu1(c[r] + bias));
      }
    }
    __syncthreads();

    // ---- layer2: [16x128] @ [128x8] tail (VALU; ~5% of total FLOPs) ----
    const int obase = half * 4;         // each lane: row mrow, 4 outputs
    float acc2[4];
#pragma unroll
    for (int q = 0; q < 4; ++q) acc2[q] = sb2[obase + q];
    for (int n = 0; n < 128; n += 2) {
      const unsigned int hp = *(const unsigned int*)&stg[mrow * STG + n];
      const float h0 = bf2f((unsigned short)(hp & 0xFFFFu));
      const float h1 = bf2f((unsigned short)(hp >> 16));
#pragma unroll
      for (int q = 0; q < 4; ++q) {
        const f32x2 w = *(const f32x2*)&sW2t[(obase + q) * 128 + n];
        acc2[q] += h0 * w[0] + h1 * w[1];
      }
    }

    if (valid) {
      if (MODE == 1) {
#pragma unroll
        for (int q = 0; q < 4; ++q)
          atomicAdd(&out[dst * 8 + obase + q], acc2[q] + addv[dst * 8 + obase + q]);
      } else {
#pragma unroll
        for (int q = 0; q < 4; ++q)
          out[(long)row * 8 + obase + q] = acc2[q];
      }
    }
  }
}

// ---------------------------------------------------------------------------
// Weight prep: transpose + bf16 conversion (one launch per MLP; 10 total).
// ---------------------------------------------------------------------------
__global__ void prep_mlp_kernel(const float* __restrict__ W0, int d_in,
                                const float* __restrict__ W1,
                                const float* __restrict__ W2,
                                unsigned short* __restrict__ W0t,
                                unsigned short* __restrict__ W1t,
                                float* __restrict__ W2t)
{
  const int i = blockIdx.x * blockDim.x + threadIdx.x;
  if (i < 128 * 32) {
    const int n = i / 32, k = i % 32;
    W0t[i] = f2bf((k < d_in) ? W0[k * 128 + n] : 0.f);
  }
  if (i < 128 * 128) {
    const int n = i / 128, k = i % 128;
    W1t[i] = f2bf(W1[k * 128 + n]);
  }
  if (i < 8 * 128) {
    const int o = i / 128, k = i % 128;
    W2t[i] = W2[k * 8 + o];
  }
}

// ---------------------------------------------------------------------------
// Small helper kernels (resource layers, scatters).
// ---------------------------------------------------------------------------
__global__ void k_zero(float* __restrict__ p, int n) {
  const int i = blockIdx.x * blockDim.x + threadIdx.x;
  if (i < n) p[i] = 0.f;
}

__global__ void k_res_lin(const float* __restrict__ h, int d_in,
                          const float* __restrict__ W, const float* __restrict__ b,
                          float* __restrict__ z, int n) {
  const int i = blockIdx.x * blockDim.x + threadIdx.x;
  if (i >= n * 8) return;
  const int node = i / 8, o = i % 8;
  float s = b[o];
  for (int k = 0; k < d_in; ++k) s += h[node * d_in + k] * W[k * 8 + o];
  z[i] = s;
}

__global__ void k_res_scatter(const float* __restrict__ z,
                              const int* __restrict__ src, const int* __restrict__ dst,
                              float* __restrict__ s, float* __restrict__ deg, int ne) {
  const int i = blockIdx.x * blockDim.x + threadIdx.x;
  if (i >= ne * 8) return;
  const int e = i / 8, o = i % 8;
  const int d = dst[e], sr = src[e];
  atomicAdd(&s[d * 8 + o], z[sr * 8 + o]);
  if (o == 0) atomicAdd(&deg[d], 1.f);
}

__global__ void k_res_div(const float* __restrict__ s, const float* __restrict__ deg,
                          float* __restrict__ h, int n) {
  const int i = blockIdx.x * blockDim.x + threadIdx.x;
  if (i >= n * 8) return;
  float d = deg[i / 8];
  d = d > 1.f ? d : 1.f;
  h[i] = s[i] / d;
}

__global__ void k_mo_scatter(const float* __restrict__ res_emb,
                             const int* __restrict__ e_op, const int* __restrict__ e_res,
                             float* __restrict__ agg, int ne) {
  const int i = blockIdx.x * blockDim.x + threadIdx.x;
  if (i >= ne * 8) return;
  const int e = i / 8, o = i % 8;
  atomicAdd(&agg[e_op[e] * 8 + o], res_emb[e_res[e] * 8 + o]);
}

// ---------------------------------------------------------------------------
// Host driver
// ---------------------------------------------------------------------------
static inline int cdiv_i(int a, int b) { return (a + b - 1) / b; }
static inline int min_i(int a, int b) { return a < b ? a : b; }

#define PERSIST_GRID 1024

extern "C" void kernel_launch(void* const* d_in, const int* in_sizes, int n_in,
                              void* d_out, int out_size, void* d_ws, size_t ws_size,
                              hipStream_t stream)
{
  (void)n_in; (void)out_size; (void)ws_size;

  const float* x_op    = (const float*)d_in[0];
  const float* x_res   = (const float*)d_in[1];
  const int*   e_prec  = (const int*)d_in[2];     // [2, EP] : row0=src, row1=dst
  const int*   e_mo_op = (const int*)d_in[3];
  const int*   e_mo_res= (const int*)d_in[4];
  const int*   e_res   = (const int*)d_in[5];     // [2, ER]

  const int n_op  = in_sizes[0] / 6;
  const int n_res = in_sizes[1] / 3;
  const int ep    = in_sizes[2] / 2;
  const int emo   = in_sizes[3];
  const int er    = in_sizes[5] / 2;

  // param leaves (jax tree order): params{'op','res'} sorted -> op first.
  // op = 2 layers, each dict sorted: comb, pred, resm, same, succ;
  // each MLP dict sorted: W0, W1, W2, b0, b1, b2.  res layers: W, b.
  const float* PW[10][6];
  for (int l = 0; l < 2; ++l)
    for (int m = 0; m < 5; ++m)
      for (int t = 0; t < 6; ++t)
        PW[l * 5 + m][t] = (const float*)d_in[6 + (l * 5 + m) * 6 + t];
  const float* resW[2] = { (const float*)d_in[66], (const float*)d_in[68] };
  const float* resB[2] = { (const float*)d_in[67], (const float*)d_in[69] };
  // mlp indices within a layer: 0=comb(24), 1=pred, 2=resm(8), 3=same, 4=succ

  // ---- carve workspace ----
  char* wp = (char*)d_ws;
  auto carve = [&](size_t bytes) -> void* {
    void* p = (void*)wp;
    wp += (bytes + 255) & ~(size_t)255;
    return p;
  };
  float* res_z  = (float*)carve((size_t)n_res * 8 * 4);
  float* res_s  = (float*)carve((size_t)n_res * 8 * 4);
  float* res_h  = (float*)carve((size_t)n_res * 8 * 4);
  float* deg    = (float*)carve((size_t)n_res * 4);
  float* agg    = (float*)carve((size_t)n_op * 8 * 4);
  float* pred_o = (float*)carve((size_t)n_op * 8 * 4);
  float* succ_o = (float*)carve((size_t)n_op * 8 * 4);
  float* resm_o = (float*)carve((size_t)n_op * 8 * 4);
  float* same_o = (float*)carve((size_t)n_op * 8 * 4);
  float* xbuf   = (float*)carve((size_t)n_op * 8 * 4);
  unsigned short* w0t[10]; unsigned short* w1t[10]; float* w2t[10];
  for (int i = 0; i < 10; ++i) {
    w0t[i] = (unsigned short*)carve(128 * 32 * 2);
    w1t[i] = (unsigned short*)carve(128 * 128 * 2);
    w2t[i] = (float*)carve(8 * 128 * 4);
  }

  float* op_emb_out  = (float*)d_out;                          // [n_op, 8]
  float* res_emb_out = (float*)d_out + (size_t)n_op * 8;       // [n_res, 8]

  // ---- weight prep (transpose + bf16) ----
  for (int l = 0; l < 2; ++l)
    for (int m = 0; m < 5; ++m) {
      const int idx  = l * 5 + m;
      const int din  = (m == 0) ? 24 : ((m == 2) ? 8 : (l == 0 ? 6 : 8));
      prep_mlp_kernel<<<64, 256, 0, stream>>>(PW[idx][0], din, PW[idx][1], PW[idx][2],
                                              w0t[idx], w1t[idx], w2t[idx]);
    }

  // ---- resource embedding: 2 x (linear + mean scatter) ----
  const int* r_src = e_res;
  const int* r_dst = e_res + er;
  for (int l = 0; l < 2; ++l) {
    const float* hin  = (l == 0) ? x_res : res_h;
    const int    din  = (l == 0) ? 3 : 8;
    float*       hout = (l == 0) ? res_h : res_emb_out;
    k_res_lin<<<cdiv_i(n_res * 8, 256), 256, 0, stream>>>(hin, din, resW[l], resB[l], res_z, n_res);
    k_zero<<<cdiv_i(n_res * 8, 256), 256, 0, stream>>>(res_s, n_res * 8);
    k_zero<<<cdiv_i(n_res, 256), 256, 0, stream>>>(deg, n_res);
    k_res_scatter<<<cdiv_i(er * 8, 256), 256, 0, stream>>>(res_z, r_src, r_dst, res_s, deg, er);
    k_res_div<<<cdiv_i(n_res * 8, 256), 256, 0, stream>>>(res_s, deg, hout, n_res);
  }

  // ---- agg_mach = segment_sum(res_emb[e_mo_res], e_mo_op) ----
  k_zero<<<cdiv_i(n_op * 8, 256), 256, 0, stream>>>(agg, n_op * 8);
  k_mo_scatter<<<cdiv_i(emo * 8, 256), 256, 0, stream>>>(res_emb_out, e_mo_op, e_mo_res, agg, emo);

  // ---- operation layers ----
  const int* p_src = e_prec;
  const int* p_dst = e_prec + ep;
  const int  ntile_n = cdiv_i(n_op, ROWS_PER_BLOCK);
  const int  ntile_e = cdiv_i(ep, ROWS_PER_BLOCK);
  const int  grid_n  = min_i(ntile_n, PERSIST_GRID);
  const int  grid_e  = min_i(ntile_e, PERSIST_GRID);

  auto node_mlp = [&](const float* xin, int din, int idx, float* outp) {
    if (din == 6)
      mlp16_kernel<0, 6><<<grid_n, MLP_BLOCK, 0, stream>>>(
          xin, nullptr, nullptr, nullptr, nullptr, nullptr, nullptr, n_op, ntile_n,
          w0t[idx], PW[idx][3], w1t[idx], PW[idx][4], w2t[idx], PW[idx][5], outp);
    else
      mlp16_kernel<0, 8><<<grid_n, MLP_BLOCK, 0, stream>>>(
          xin, nullptr, nullptr, nullptr, nullptr, nullptr, nullptr, n_op, ntile_n,
          w0t[idx], PW[idx][3], w1t[idx], PW[idx][4], w2t[idx], PW[idx][5], outp);
  };

  const float* xcur = x_op;
  int dcur = 6;
  for (int l = 0; l < 2; ++l) {
    const int mb = l * 5;
    // node MLPs (dense)
    node_mlp(xcur, dcur, mb + 1, pred_o);
    node_mlp(xcur, dcur, mb + 4, succ_o);
    node_mlp(xcur, dcur, mb + 3, same_o);
    node_mlp(agg, 8, mb + 2, resm_o);
    // per-edge comb MLP + same[dst], scatter-add by dst
    float* outp = (l == 0) ? xbuf : op_emb_out;
    k_zero<<<cdiv_i(n_op * 8, 256), 256, 0, stream>>>(outp, n_op * 8);
    mlp16_kernel<1, 24><<<grid_e, MLP_BLOCK, 0, stream>>>(
        nullptr, resm_o, pred_o, succ_o, p_src, p_dst, same_o, ep, ntile_e,
        w0t[mb + 0], PW[mb + 0][3], w1t[mb + 0], PW[mb + 0][4], w2t[mb + 0], PW[mb + 0][5], outp);
    xcur = outp;
    dcur = 8;
  }
}
